// FeatMatchNN_32865089749665
// MI455X (gfx1250) — matrix operand order repaired and checked by reading, not compiled, not run on hardware
//
#include <hip/hip_runtime.h>
#include <hip/hip_bf16.h>
#include <math.h>

typedef __attribute__((ext_vector_type(16))) __bf16 v16bf;
typedef __attribute__((ext_vector_type(8)))  __bf16 v8bf;
typedef __attribute__((ext_vector_type(8)))  float  v8f;

#define NEG_SLOPE 0.01f
#define LPTS 256

__device__ __forceinline__ float leaky(float x) {
    return x > 0.f ? x : NEG_SLOPE * x;
}

// ---------------------------------------------------------------------------
// Encoder direct conv (small channel counts) + leaky ReLU, 'same' padding.
// in: (Ci,H,W)  w: (Co,Ci,K,K)  out: (Co,H,W)
// ---------------------------------------------------------------------------
__global__ void conv2d_leaky_k(const float* __restrict__ in, const float* __restrict__ w,
                               const float* __restrict__ bias, float* __restrict__ out,
                               int Ci, int Co, int H, int W, int K) {
    size_t total = (size_t)Co * H * W;
    size_t idx = (size_t)blockIdx.x * blockDim.x + threadIdx.x;
    if (idx >= total) return;
    int x = idx % W;
    size_t r = idx / W;
    int y = r % H;
    int co = r / H;
    int p = K >> 1;
    float acc = bias[co];
    for (int ci = 0; ci < Ci; ++ci) {
        const float* ip = in + (size_t)ci * H * W;
        const float* wp = w + ((size_t)co * Ci + ci) * K * K;
        for (int ky = 0; ky < K; ++ky) {
            int iy = y + ky - p;
            if (iy < 0 || iy >= H) continue;
            const float* iprow = ip + (size_t)iy * W;
            const float* wrow = wp + (size_t)ky * K;
            for (int kx = 0; kx < K; ++kx) {
                int ix = x + kx - p;
                if (ix < 0 || ix >= W) continue;
                acc = fmaf(iprow[ix], wrow[kx], acc);
            }
        }
    }
    out[idx] = leaky(acc);
}

// ---------------------------------------------------------------------------
// Pack h2 (C,HW) fp32 -> (HW, CP) bf16, zero-padded K. Coalesced writes.
// ---------------------------------------------------------------------------
template <int C>
__global__ void pack_bf16_k(const float* __restrict__ src, __bf16* __restrict__ dst, int HW) {
    constexpr int CP = (C + 31) & ~31;
    size_t total = (size_t)HW * CP;
    size_t idx = (size_t)blockIdx.x * blockDim.x + threadIdx.x;
    if (idx >= total) return;
    int c = idx % CP;
    size_t hw = idx / CP;
    dst[idx] = (c < C) ? (__bf16)src[(size_t)c * HW + hw] : (__bf16)0.0f;
}

// ---------------------------------------------------------------------------
// Gather point features directly into padded bf16 (L, CP):
// Q[l,c] = h1[c, (p0>>s)*Wi + (p1>>s)]
// ---------------------------------------------------------------------------
template <int C>
__global__ void gather_q_k(const float* __restrict__ h1, const int* __restrict__ pts,
                           __bf16* __restrict__ q, int HW, int Wi, int shift) {
    constexpr int CP = (C + 31) & ~31;
    int idx = blockIdx.x * blockDim.x + threadIdx.x;
    if (idx >= LPTS * CP) return;
    int c = idx % CP;
    int l = idx / CP;
    if (c < C) {
        int p0 = pts[l * 2 + 0] >> shift;
        int p1 = pts[l * 2 + 1] >> shift;
        q[idx] = (__bf16)h1[(size_t)c * HW + (size_t)p0 * Wi + p1];
    } else {
        q[idx] = (__bf16)0.0f;
    }
}

// ---------------------------------------------------------------------------
// Attention scores via WMMA bf16: s[hw, l] = sum_c h2[c,hw] * Q[c,l].
// Operands pre-staged padded bf16: A = (HW, CP), B = (L, CP).
// One wave computes a 16-row pixel block against ALL 16 L-tiles, holding the
// A fragment in registers (one contiguous 32B load per K-chunk:
// lanes 0-15 K 0..15, lanes 16-31 K 16..31). B fragment is one contiguous 32B load per chunk
// (lanes 0-15 K 0..15, lanes 16-31 K 16..31). Output (HW, L) row-major ==
// flat (h,w,l) order, preserving the reference's reinterpret-reshape.
// ---------------------------------------------------------------------------
template <int C>
__global__ void attn_scores_wmma_k(const __bf16* __restrict__ a2,  // (HW, CP)
                                   const __bf16* __restrict__ q,   // (LPTS, CP)
                                   float* __restrict__ s, int HW) {
    constexpr int CP = (C + 31) & ~31;
    constexpr int NCH = CP / 32;
    int lane = threadIdx.x & 31;
    int wave = threadIdx.x >> 5;
    bool lo = lane < 16;
    int sub = lane & 15;
    int pixBase = blockIdx.x * 128 + wave * 16;

    const __bf16* arow = a2 + (size_t)(pixBase + sub) * CP;
    v16bf A[NCH];
#pragma unroll
    for (int ch = 0; ch < NCH; ++ch) {
        A[ch] = *(const v16bf*)(arow + ch * 32 + (lo ? 0 : 16));
    }

    int mOff = lo ? 0 : 8;
#pragma unroll
    for (int lt = 0; lt < LPTS / 16; ++lt) {
        const __bf16* qrow = q + (size_t)(lt * 16 + sub) * CP + (lo ? 0 : 16);
        v8f acc = {};
#pragma unroll
        for (int ch = 0; ch < NCH; ++ch) {
            v16bf B = *(const v16bf*)(qrow + ch * 32);
            acc = __builtin_amdgcn_wmma_f32_16x16x32_bf16(
                false, A[ch], false, B, (short)0, acc, false, false);
        }
#pragma unroll
        for (int r = 0; r < 8; ++r)
            s[(size_t)(pixBase + r + mOff) * LPTS + lt * 16 + sub] = acc[r];
    }
}

// ---------------------------------------------------------------------------
// In-place softmax over rows of length rowLen (flat reinterpreted view).
// 1024 threads per row.
// ---------------------------------------------------------------------------
__global__ void softmax_rows_k(float* __restrict__ data, int rowLen) {
    float* p = data + (size_t)blockIdx.x * rowLen;
    __shared__ float red[1024];
    int t = threadIdx.x;
    float m = -INFINITY;
    for (int i = t; i < rowLen; i += 1024) m = fmaxf(m, p[i]);
    red[t] = m;
    __syncthreads();
    for (int st = 512; st > 0; st >>= 1) {
        if (t < st) red[t] = fmaxf(red[t], red[t + st]);
        __syncthreads();
    }
    m = red[0];
    __syncthreads();
    float sum = 0.f;
    for (int i = t; i < rowLen; i += 1024) {
        float e = expf(p[i] - m);
        p[i] = e;
        sum += e;
    }
    red[t] = sum;
    __syncthreads();
    for (int st = 512; st > 0; st >>= 1) {
        if (t < st) red[t] += red[t + st];
        __syncthreads();
    }
    float inv = 1.0f / red[0];
    for (int i = t; i < rowLen; i += 1024) p[i] *= inv;
}

// ---------------------------------------------------------------------------
// 2x2 max pool. in: (C,H,W) out: (C,H/2,W/2)
// ---------------------------------------------------------------------------
__global__ void maxpool2_k(const float* __restrict__ in, float* __restrict__ out,
                           int C, int H, int W) {
    int Ho = H >> 1, Wo = W >> 1;
    size_t total = (size_t)C * Ho * Wo;
    size_t idx = (size_t)blockIdx.x * blockDim.x + threadIdx.x;
    if (idx >= total) return;
    int x = idx % Wo;
    size_t r = idx / Wo;
    int y = r % Ho;
    int c = r / Ho;
    const float* ip = in + ((size_t)c * H + 2 * y) * W + 2 * x;
    out[idx] = fmaxf(fmaxf(ip[0], ip[1]), fmaxf(ip[W], ip[W + 1]));
}

// ---------------------------------------------------------------------------
// Transposed conv tap: sum over Ci of in[ci, iy, ix] * w[ci, co, ky, kx]
// ---------------------------------------------------------------------------
template <int Ci>
__device__ __forceinline__ float tapsum(const float* __restrict__ inb,
                                        const float* __restrict__ w,
                                        int Co, int co, int HinWin,
                                        int off, int ky, int kx) {
    const float* ip = inb + off;
    const float* wp = w + (size_t)co * 9 + ky * 3 + kx;
    float a = 0.f;
#pragma unroll
    for (int ci = 0; ci < Ci; ++ci)
        a = fmaf(ip[(size_t)ci * HinWin], wp[(size_t)ci * Co * 9], a);
    return a;
}

// ---------------------------------------------------------------------------
// ConvTranspose2d(k=3, s=2, p=1, op=1): out (N,Co,2H,2W), w: (Ci,Co,3,3).
// blockIdx.y = output parity class (py,px) -> uniform tap structure, no lane
// divergence. Optionally fused per-batch attention multiply and leaky.
// oy = 2*iy - 1 + ky  =>  parity py: py==0 -> (ky=1,dy=0); py==1 -> (ky=0,
// dy=1),(ky=2,dy=0). Same in x.
// ---------------------------------------------------------------------------
template <int Ci, bool HASATTN, bool LEAKY>
__global__ void convt_fused_k(const float* __restrict__ in, const float* __restrict__ w,
                              const float* __restrict__ bias, const float* __restrict__ attn,
                              float* __restrict__ out, int N, int Co, int Hin, int Win) {
    int py = blockIdx.y >> 1;
    int px = blockIdx.y & 1;
    size_t total = (size_t)N * Co * Hin * Win;
    size_t idx = (size_t)blockIdx.x * blockDim.x + threadIdx.x;
    if (idx >= total) return;
    int x = idx % Win;
    size_t r = idx / Win;
    int y = r % Hin;
    r /= Hin;
    int co = r % Co;
    int n = r / Co;
    int Ho = Hin * 2, Wo = Win * 2;
    int oy = 2 * y + py, ox = 2 * x + px;
    int HW = Hin * Win;
    const float* inb = in + (size_t)n * Ci * HW;

    float acc = bias[co];
    if (py == 0) {
        if (px == 0) {
            acc += tapsum<Ci>(inb, w, Co, co, HW, y * Win + x, 1, 1);
        } else {
            if (x + 1 < Win) acc += tapsum<Ci>(inb, w, Co, co, HW, y * Win + x + 1, 1, 0);
            acc += tapsum<Ci>(inb, w, Co, co, HW, y * Win + x, 1, 2);
        }
    } else {
        if (px == 0) {
            if (y + 1 < Hin) acc += tapsum<Ci>(inb, w, Co, co, HW, (y + 1) * Win + x, 0, 1);
            acc += tapsum<Ci>(inb, w, Co, co, HW, y * Win + x, 2, 1);
        } else {
            if (y + 1 < Hin) {
                if (x + 1 < Win) acc += tapsum<Ci>(inb, w, Co, co, HW, (y + 1) * Win + x + 1, 0, 0);
                acc += tapsum<Ci>(inb, w, Co, co, HW, (y + 1) * Win + x, 0, 2);
            }
            if (x + 1 < Win) acc += tapsum<Ci>(inb, w, Co, co, HW, y * Win + x + 1, 2, 0);
            acc += tapsum<Ci>(inb, w, Co, co, HW, y * Win + x, 2, 2);
        }
    }
    if (HASATTN) acc *= attn[(size_t)n * Ho * Wo + (size_t)oy * Wo + ox];
    if (LEAKY) acc = leaky(acc);
    out[(((size_t)n * Co + co) * Ho + oy) * Wo + ox] = acc;
}

// ---------------------------------------------------------------------------
// dec0 batch-broadcast: out[n,c,hw] = leaky(tmp[c,hw] * attn[n,hw])
// ---------------------------------------------------------------------------
__global__ void bcast_mul_leaky_k(const float* __restrict__ tmp, const float* __restrict__ attn,
                                  float* __restrict__ out, int C, int HW) {
    size_t total = (size_t)LPTS * C * HW;
    size_t idx = (size_t)blockIdx.x * blockDim.x + threadIdx.x;
    if (idx >= total) return;
    int hw = idx % HW;
    size_t r = idx / HW;
    int c = r % C;
    int n = r / C;
    out[idx] = leaky(tmp[(size_t)c * HW + hw] * attn[(size_t)n * HW + hw]);
}

// ---------------------------------------------------------------------------
// valid[l] = (max over H*W of conf[l,:,:] > 1) as 0/1 float
// ---------------------------------------------------------------------------
__global__ void valid_k(const float* __restrict__ conf, float* __restrict__ out, int HW) {
    const float* p = conf + (size_t)blockIdx.x * HW;
    __shared__ float red[256];
    int t = threadIdx.x;
    float m = -INFINITY;
    for (int i = t; i < HW; i += 256) m = fmaxf(m, p[i]);
    red[t] = m;
    __syncthreads();
    for (int st = 128; st > 0; st >>= 1) {
        if (t < st) red[t] = fmaxf(red[t], red[t + st]);
        __syncthreads();
    }
    if (t == 0) out[blockIdx.x] = (red[0] > 1.0f) ? 1.0f : 0.0f;
}

// ---------------------------------------------------------------------------
// Workspace layout (float units)
// ---------------------------------------------------------------------------
static constexpr size_t NBUF      = 524288;                       // 8*256*256 max conv buf
static constexpr size_t OFF_H1A   = 0;
static constexpr size_t OFF_H1B   = OFF_H1A + NBUF;
static constexpr size_t OFF_H2A   = OFF_H1B + NBUF;
static constexpr size_t OFF_H2B   = OFF_H2A + NBUF;
static constexpr size_t OFF_ATTN0 = OFF_H2B + NBUF;               // 256*65536
static constexpr size_t OFF_ATTN1 = OFF_ATTN0 + 256ull * 65536;   // 256*16384
static constexpr size_t OFF_ATTN2 = OFF_ATTN1 + 256ull * 16384;   // 256*4096
static constexpr size_t OFF_ATTN3 = OFF_ATTN2 + 256ull * 4096;    // 256*1024
static constexpr size_t OFF_TMP0  = OFF_ATTN3 + 256ull * 1024;    // 32*32*32
static constexpr size_t OFF_DECA  = OFF_TMP0 + 32768;             // 256*32*32*32
static constexpr size_t OFF_DECB  = OFF_DECA + 256ull * 32 * 1024;// 256*16*64*64
static constexpr size_t OFF_DECC  = OFF_DECB + 256ull * 16 * 4096;// 256*8*128*128
static constexpr size_t OFF_PACK  = OFF_DECC + 256ull * 8 * 16384;// bf16 (HW,CP): <= 65536*32 bf16
static constexpr size_t OFF_QBF   = OFF_PACK + 2097152;           // bf16 (256,64)
static constexpr size_t CONF_ELEMS = 256ull * 256 * 256;          // (1,L,H,W)

static inline dim3 gridFor(size_t total, int block = 256) {
    return dim3((unsigned)((total + block - 1) / block));
}

// Per-level attention path (templated on channel count)
template <int C>
static void attention_level(const float* h1a, const float* h2a, const int* pts,
                            __bf16* h2p, __bf16* qbf, float* attnBuf,
                            int Hc, int lvl, hipStream_t stream) {
    constexpr int CP = (C + 31) & ~31;
    int HW = Hc * Hc;
    gather_q_k<C><<<gridFor((size_t)LPTS * CP), 256, 0, stream>>>(h1a, pts, qbf, HW, Hc, lvl);
    pack_bf16_k<C><<<gridFor((size_t)HW * CP), 256, 0, stream>>>(h2a, h2p, HW);
    attn_scores_wmma_k<C><<<dim3(HW / 128), 256, 0, stream>>>(h2p, qbf, attnBuf, HW);
    softmax_rows_k<<<LPTS, 1024, 0, stream>>>(attnBuf, HW);
}

extern "C" void kernel_launch(void* const* d_in, const int* in_sizes, int n_in,
                              void* d_out, int out_size, void* d_ws, size_t ws_size,
                              hipStream_t stream) {
    (void)in_sizes; (void)n_in; (void)out_size; (void)ws_size;
    const float* x1  = (const float*)d_in[0];
    const float* x2  = (const float*)d_in[1];
    const int*   pts = (const int*)d_in[2];
    const float* ew[4] = {(const float*)d_in[3], (const float*)d_in[5],
                          (const float*)d_in[7], (const float*)d_in[9]};
    const float* eb[4] = {(const float*)d_in[4], (const float*)d_in[6],
                          (const float*)d_in[8], (const float*)d_in[10]};
    const float* dw[4] = {(const float*)d_in[11], (const float*)d_in[13],
                          (const float*)d_in[15], (const float*)d_in[17]};
    const float* db[4] = {(const float*)d_in[12], (const float*)d_in[14],
                          (const float*)d_in[16], (const float*)d_in[18]};

    float* ws = (float*)d_ws;
    float* h1a = ws + OFF_H1A;
    float* h1b = ws + OFF_H1B;
    float* h2a = ws + OFF_H2A;
    float* h2b = ws + OFF_H2B;
    float* attn[4] = {ws + OFF_ATTN0, ws + OFF_ATTN1, ws + OFF_ATTN2, ws + OFF_ATTN3};
    __bf16* h2p = (__bf16*)(ws + OFF_PACK);
    __bf16* qbf = (__bf16*)(ws + OFF_QBF);

    const int Ci_[4] = {3, 8, 16, 32};
    const int Co_[4] = {8, 16, 32, 64};
    const int K_[4]  = {3, 5, 7, 9};
    const int Hs[4]  = {256, 128, 64, 32};

    const float* in1 = x1;
    const float* in2 = x2;
    for (int lvl = 0; lvl < 4; ++lvl) {
        int Hc = Hs[lvl], Wc = Hc, HW = Hc * Wc;
        int Ci = Ci_[lvl], Co = Co_[lvl], K = K_[lvl];
        size_t convTotal = (size_t)Co * HW;
        conv2d_leaky_k<<<gridFor(convTotal), 256, 0, stream>>>(in1, ew[lvl], eb[lvl], h1a, Ci, Co, Hc, Wc, K);
        conv2d_leaky_k<<<gridFor(convTotal), 256, 0, stream>>>(in2, ew[lvl], eb[lvl], h2a, Ci, Co, Hc, Wc, K);
        switch (lvl) {
            case 0: attention_level<8>(h1a, h2a, pts, h2p, qbf, attn[0], Hc, lvl, stream); break;
            case 1: attention_level<16>(h1a, h2a, pts, h2p, qbf, attn[1], Hc, lvl, stream); break;
            case 2: attention_level<32>(h1a, h2a, pts, h2p, qbf, attn[2], Hc, lvl, stream); break;
            case 3: attention_level<64>(h1a, h2a, pts, h2p, qbf, attn[3], Hc, lvl, stream); break;
        }
        size_t poolTotal = (size_t)Co * (HW / 4);
        maxpool2_k<<<gridFor(poolTotal), 256, 0, stream>>>(h1a, h1b, Co, Hc, Wc);
        maxpool2_k<<<gridFor(poolTotal), 256, 0, stream>>>(h2a, h2b, Co, Hc, Wc);
        in1 = h1b;
        in2 = h2b;
    }

    // Decoder. h2b = (64,16,16).
    float* tmp0 = ws + OFF_TMP0;
    float* decA = ws + OFF_DECA;
    float* decB = ws + OFF_DECB;
    float* decC = ws + OFF_DECC;
    float* conf = (float*)d_out;

    // dec0: convT once (batch 1), then broadcast-multiply by attn3 + leaky
    convt_fused_k<64, false, false><<<dim3((unsigned)((32ull * 16 * 16 + 255) / 256), 4), 256, 0, stream>>>(
        h2b, dw[0], db[0], nullptr, tmp0, 1, 32, 16, 16);
    bcast_mul_leaky_k<<<gridFor(256ull * 32 * 1024), 256, 0, stream>>>(tmp0, attn[3], decA, 32, 1024);
    // dec1: (256,32,32,32) -> (256,16,64,64)
    convt_fused_k<32, true, true><<<dim3((unsigned)((256ull * 16 * 32 * 32 + 255) / 256), 4), 256, 0, stream>>>(
        decA, dw[1], db[1], attn[2], decB, 256, 16, 32, 32);
    // dec2: (256,16,64,64) -> (256,8,128,128)
    convt_fused_k<16, true, true><<<dim3((unsigned)((256ull * 8 * 64 * 64 + 255) / 256), 4), 256, 0, stream>>>(
        decB, dw[2], db[2], attn[1], decC, 256, 8, 64, 64);
    // dec3: (256,8,128,128) -> (256,1,256,256) == conf, no leaky
    convt_fused_k<8, true, false><<<dim3((unsigned)((256ull * 1 * 128 * 128 + 255) / 256), 4), 256, 0, stream>>>(
        decC, dw[3], db[3], attn[0], conf, 256, 1, 128, 128);
    // valid flags after conf
    valid_k<<<LPTS, 256, 0, stream>>>(conf, conf + CONF_ELEMS, 65536);
}
